// EvalSparseMoeBlock_45896020525335
// MI455X (gfx1250) — matrix operand order repaired and checked by reading, not compiled, not run on hardware
//
#include <hip/hip_runtime.h>
#include <hip/hip_bf16.h>

// ---------------------------------------------------------------------------
// MoE block for MI455X (gfx1250): top-2 gathered dispatch + bf16 WMMA GEMMs,
// software-pipelined K loop, async global->LDS staging for bf16 A tiles.
// T=4096 tokens, H=1024, F=4096, E=8.
// ---------------------------------------------------------------------------

typedef __bf16 v16bf __attribute__((ext_vector_type(16)));
typedef float  v8f   __attribute__((ext_vector_type(8)));

constexpr int T = 4096;   // B*S
constexpr int H = 1024;
constexpr int F = 4096;
constexpr int E = 8;

// --------------------------- async-to-LDS probe ----------------------------
#ifndef CDNA5_TRY_ASYNC_LDS
#define CDNA5_TRY_ASYNC_LDS 1
#endif

#if CDNA5_TRY_ASYNC_LDS && defined(__has_builtin)
#if __has_builtin(__builtin_amdgcn_global_load_async_to_lds_b32)
#define HAS_ASYNC_LDS 1
#endif
#endif
#ifndef HAS_ASYNC_LDS
#define HAS_ASYNC_LDS 0
#endif

#if HAS_ASYNC_LDS
typedef __attribute__((address_space(1))) int as1_int;
typedef __attribute__((address_space(3))) int as3_int;

__device__ __forceinline__ void async_copy_b32(const void* g, void* l) {
  // AS1 generic address == global address; AS3 pointer is the 32-bit LDS
  // offset (low half of the generic LDS address).
  as1_int* gp = (as1_int*)(unsigned long long)(uintptr_t)g;
  as3_int* lp = (as3_int*)(unsigned int)(uintptr_t)l;
  __builtin_amdgcn_global_load_async_to_lds_b32(gp, lp, 0, 0);
}
__device__ __forceinline__ void wait_async0() {
#if __has_builtin(__builtin_amdgcn_s_wait_asynccnt)
  __builtin_amdgcn_s_wait_asynccnt(0);
#else
  asm volatile("s_wait_asynccnt 0x0" ::: "memory");
#endif
}
#endif  // HAS_ASYNC_LDS

// f32 -> bf16 with round-to-nearest-even
__device__ __forceinline__ unsigned int f2bf1(float f) {
  unsigned int u = __float_as_uint(f);
  u += 0x7FFFu + ((u >> 16) & 1u);
  return u >> 16;
}
__device__ __forceinline__ unsigned int f2bf_pk(float lo, float hi) {
  return f2bf1(lo) | (f2bf1(hi) << 16);
}
__device__ __forceinline__ v8f zero8() {
  v8f z;
#pragma unroll
  for (int i = 0; i < 8; ++i) z[i] = 0.f;
  return z;
}

// ---------------------------------------------------------------------------
// x (f32) -> xb (bf16), packed pair stores
// ---------------------------------------------------------------------------
__global__ __launch_bounds__(256) void cvt_bf16_kernel(
    const float* __restrict__ x, unsigned int* __restrict__ xb, int n2) {
  int i = blockIdx.x * 256 + threadIdx.x;
  if (i < n2) xb[i] = f2bf_pk(x[2 * i], x[2 * i + 1]);
}

// ---------------------------------------------------------------------------
// Router: one wave32 per token. softmax(x @ w_gate), top-2, append
// (token, prob) into per-expert compacted lists (order-invariant result).
// ---------------------------------------------------------------------------
__global__ __launch_bounds__(256) void moe_router_kernel(
    const float* __restrict__ x, const float* __restrict__ wg,
    int* __restrict__ idx, float* __restrict__ wgt, int* __restrict__ cnt) {
  const int gid  = blockIdx.x * 256 + threadIdx.x;
  const int tok  = gid >> 5;
  const int lane = gid & 31;
  if (tok >= T) return;

  const float* xr = x + (size_t)tok * H;
  float acc[E];
#pragma unroll
  for (int e = 0; e < E; ++e) acc[e] = 0.f;

  for (int h = lane; h < H; h += 32) {
    float xv = xr[h];
    const float* wr = wg + (size_t)h * E;
#pragma unroll
    for (int e = 0; e < E; ++e) acc[e] += xv * wr[e];
  }
#pragma unroll
  for (int off = 16; off > 0; off >>= 1) {
#pragma unroll
    for (int e = 0; e < E; ++e) acc[e] += __shfl_xor(acc[e], off, 32);
  }

  if (lane == 0) {
    float mx = acc[0];
#pragma unroll
    for (int e = 1; e < E; ++e) mx = fmaxf(mx, acc[e]);
    float p[E];
    float s = 0.f;
#pragma unroll
    for (int e = 0; e < E; ++e) { p[e] = __expf(acc[e] - mx); s += p[e]; }
    const float inv = 1.f / s;  // softmax probs (top-k NOT renormalized)

    int b1 = 0;
#pragma unroll
    for (int e = 1; e < E; ++e) if (p[e] > p[b1]) b1 = e;
    int b2 = (b1 == 0) ? 1 : 0;
#pragma unroll
    for (int e = 0; e < E; ++e) if (e != b1 && p[e] > p[b2]) b2 = e;

    int pos1 = atomicAdd(&cnt[b1], 1);
    idx[b1 * T + pos1] = tok;
    wgt[b1 * T + pos1] = p[b1] * inv;
    int pos2 = atomicAdd(&cnt[b2], 1);
    idx[b2 * T + pos2] = tok;
    wgt[b2 * T + pos2] = p[b2] * inv;
  }
}

// ---------------------------------------------------------------------------
// LDS fragment swizzles: each WMMA fragment becomes one contiguous 32-byte
// per-lane LDS read.
// A 16x32 bf16 fragment: lane = m + 16*half; elem h -> K per ISA §7.12.2.
// B 32x16 bf16 fragment: lane = n + 16*((k>>3)&1); elem = 8*(k>>4) + k%8.
// ---------------------------------------------------------------------------
__device__ __forceinline__ int a_swz(int row, int k) {  // row 0..63, k 0..31
  int grp = k >> 3, half = grp & 1;
  int hh = ((grp >> 1) << 3) + (k & 7);
  return ((row >> 4) << 9) + (((row & 15) + (half << 4)) << 4) + hh;
}
__device__ __forceinline__ int b_swz(int k, int n) {    // k 0..31, n 0..127
  return ((n >> 4) << 9) + (((n & 15) + (((k >> 3) & 1) << 4)) << 4) + (((k >> 4) << 3) + (k & 7));
}

// ---------------------------------------------------------------------------
// GEMM1 fused gate+up: h[slot,n] = silu(x@Wg) * (x@Wu) -> bf16.
// Block tile 64x128, K-step 32, 8 waves each owning 32x32 (2x2 WMMA frags,
// two accumulator sets). K loop software-pipelined: B prefetched into
// registers (with f32->bf16 conversion); A double-buffered, staged by
// async global->LDS copies when available.
// ---------------------------------------------------------------------------
__global__ __launch_bounds__(256) void moe_gate_up_kernel(
    const unsigned short* __restrict__ xb,      // T x H bf16
    const float* __restrict__ Wg,               // H x F f32 (expert slice)
    const float* __restrict__ Wu,               // H x F f32 (expert slice)
    const int* __restrict__ idx_e,
    const int* __restrict__ cnt_e,
    unsigned short* __restrict__ hbuf) {        // T x F bf16
  const int cnt = *cnt_e;
  const int m0 = blockIdx.y * 64;
  if (m0 >= cnt) return;
  const int n0 = blockIdx.x * 128;

  __shared__ __align__(32) unsigned short sA[2][2048];  // 2 x (64x32)
  __shared__ __align__(32) unsigned short sBg[4096];    // 32x128
  __shared__ __align__(32) unsigned short sBu[4096];    // 32x128
  __shared__ int sTok[64];

  const int tid = threadIdx.x;
  const int lane = tid & 31;
  const int wave = tid >> 5;
  const int wm = wave & 1;
  const int wn = wave >> 1;

  if (tid < 64) {
    int s = m0 + tid;
    sTok[tid] = idx_e[s < cnt ? s : cnt - 1];  // pad rows duplicate last token
  }
  __syncthreads();

  // Per-thread staging descriptors (fully unrolled -> registers).
  int aoff[4], agoff[4];
#pragma unroll
  for (int j = 0; j < 4; ++j) {
    int q = tid + 256 * j;
    int row = q >> 4;
    int k = (q & 15) << 1;
    aoff[j]  = a_swz(row, k);
    agoff[j] = sTok[row] * H + k;  // + k0 each step
  }
  int boff[8], bgoff[8];
#pragma unroll
  for (int j = 0; j < 8; ++j) {
    int q = tid + 256 * j;
    int n = q & 127;
    int k = (q >> 7) << 1;
    boff[j]  = b_swz(k, n);
    bgoff[j] = k * F + n0 + n;     // + k0*F each step
  }

  v8f accG[2][2], accU[2][2];
#pragma unroll
  for (int i = 0; i < 2; ++i)
#pragma unroll
    for (int j = 0; j < 2; ++j) { accG[i][j] = zero8(); accU[i][j] = zero8(); }

  unsigned int pBg[8], pBu[8];
#if !HAS_ASYNC_LDS
  unsigned int pA[4];
#endif

  // Prologue: stage tile k0 = 0.
#if HAS_ASYNC_LDS
#pragma unroll
  for (int j = 0; j < 4; ++j)
    async_copy_b32(xb + agoff[j], &sA[0][aoff[j]]);
#else
#pragma unroll
  for (int j = 0; j < 4; ++j)
    pA[j] = *(const unsigned int*)(xb + agoff[j]);
#endif
#pragma unroll
  for (int j = 0; j < 8; ++j) {
    int o = bgoff[j];
    pBg[j] = f2bf_pk(Wg[o], Wg[o + F]);
    pBu[j] = f2bf_pk(Wu[o], Wu[o + F]);
  }

  for (int k0 = 0; k0 < H; k0 += 32) {
    const int cur = (k0 >> 5) & 1;
    // Commit staged tile.
#if HAS_ASYNC_LDS
    wait_async0();
#else
#pragma unroll
    for (int j = 0; j < 4; ++j)
      *(unsigned int*)(&sA[cur][aoff[j]]) = pA[j];
#endif
#pragma unroll
    for (int j = 0; j < 8; ++j) {
      *(unsigned int*)(sBg + boff[j]) = pBg[j];
      *(unsigned int*)(sBu + boff[j]) = pBu[j];
    }
    __syncthreads();

    // Prefetch next K tile (overlaps the WMMA block below).
    if (k0 + 32 < H) {
#if HAS_ASYNC_LDS
#pragma unroll
      for (int j = 0; j < 4; ++j)
        async_copy_b32(xb + agoff[j] + k0 + 32, &sA[cur ^ 1][aoff[j]]);
#else
#pragma unroll
      for (int j = 0; j < 4; ++j)
        pA[j] = *(const unsigned int*)(xb + agoff[j] + k0 + 32);
#endif
#pragma unroll
      for (int j = 0; j < 8; ++j) {
        int o = bgoff[j] + (k0 + 32) * F;
        pBg[j] = f2bf_pk(Wg[o], Wg[o + F]);
        pBu[j] = f2bf_pk(Wu[o], Wu[o + F]);
      }
    }

    v16bf a[2], bg[2], bu[2];
#pragma unroll
    for (int i = 0; i < 2; ++i)
      a[i] = *(const v16bf*)(&sA[cur][(((wm << 1) + i) << 9) + (lane << 4)]);
#pragma unroll
    for (int j = 0; j < 2; ++j) {
      bg[j] = *(const v16bf*)(sBg + (((wn << 1) + j) << 9) + (lane << 4));
      bu[j] = *(const v16bf*)(sBu + (((wn << 1) + j) << 9) + (lane << 4));
    }
#pragma unroll
    for (int i = 0; i < 2; ++i) {
#pragma unroll
      for (int j = 0; j < 2; ++j) {
        accG[i][j] = __builtin_amdgcn_wmma_f32_16x16x32_bf16(
            false, a[i], false, bg[j], (short)0, accG[i][j], false, false);
        accU[i][j] = __builtin_amdgcn_wmma_f32_16x16x32_bf16(
            false, a[i], false, bu[j], (short)0, accU[i][j], false, false);
      }
    }
    __syncthreads();
  }

  // Epilogue: h = silu(g)*u -> bf16. silu via v_rcp_f32 (no IEEE div chain).
#pragma unroll
  for (int i = 0; i < 2; ++i) {
#pragma unroll
    for (int j = 0; j < 2; ++j) {
#pragma unroll
      for (int r = 0; r < 8; ++r) {
        float g = accG[i][j][r];
        float u = accU[i][j][r];
        float hv = g * __builtin_amdgcn_rcpf(1.f + __expf(-g)) * u;
        int row = (wm << 5) + (i << 4) + r + ((lane >> 4) << 3);
        int col = n0 + (wn << 5) + (j << 4) + (lane & 15);
        hbuf[(size_t)(m0 + row) * F + col] = (unsigned short)f2bf1(hv);
      }
    }
  }
}

// ---------------------------------------------------------------------------
// GEMM2 down-proj: out[token,n] += (h @ Wd)[slot,n] * w[slot].
// Same pipelined tiling, K over F. Non-atomic accumulate is race-free:
// tokens unique within an expert; experts stream-ordered.
// ---------------------------------------------------------------------------
__global__ __launch_bounds__(256) void moe_down_kernel(
    const unsigned short* __restrict__ hbuf,    // T x F bf16
    const float* __restrict__ Wd,               // F x H f32 (expert slice)
    const int* __restrict__ idx_e,
    const float* __restrict__ wgt_e,
    const int* __restrict__ cnt_e,
    float* __restrict__ out) {                  // T x H f32
  const int cnt = *cnt_e;
  const int m0 = blockIdx.y * 64;
  if (m0 >= cnt) return;
  const int n0 = blockIdx.x * 128;

  __shared__ __align__(32) unsigned short sA[2][2048];
  __shared__ __align__(32) unsigned short sB[4096];

  const int tid = threadIdx.x;
  const int lane = tid & 31;
  const int wave = tid >> 5;
  const int wm = wave & 1;
  const int wn = wave >> 1;

  int aoff[4], agoff[4];
#pragma unroll
  for (int j = 0; j < 4; ++j) {
    int q = tid + 256 * j;
    int row = q >> 4;
    int k = (q & 15) << 1;
    aoff[j]  = a_swz(row, k);
    agoff[j] = (m0 + row) * F + k;  // + k0 each step
  }
  int boff[8], bgoff[8];
#pragma unroll
  for (int j = 0; j < 8; ++j) {
    int q = tid + 256 * j;
    int n = q & 127;
    int k = (q >> 7) << 1;
    boff[j]  = b_swz(k, n);
    bgoff[j] = k * H + n0 + n;      // + k0*H each step
  }

  v8f acc[2][2];
#pragma unroll
  for (int i = 0; i < 2; ++i)
#pragma unroll
    for (int j = 0; j < 2; ++j) acc[i][j] = zero8();

  unsigned int pB[8];
#if !HAS_ASYNC_LDS
  unsigned int pA[4];
#endif

#if HAS_ASYNC_LDS
#pragma unroll
  for (int j = 0; j < 4; ++j)
    async_copy_b32(hbuf + agoff[j], &sA[0][aoff[j]]);
#else
#pragma unroll
  for (int j = 0; j < 4; ++j)
    pA[j] = *(const unsigned int*)(hbuf + agoff[j]);
#endif
#pragma unroll
  for (int j = 0; j < 8; ++j) {
    int o = bgoff[j];
    pB[j] = f2bf_pk(Wd[o], Wd[o + H]);
  }

  for (int k0 = 0; k0 < F; k0 += 32) {
    const int cur = (k0 >> 5) & 1;
#if HAS_ASYNC_LDS
    wait_async0();
#else
#pragma unroll
    for (int j = 0; j < 4; ++j)
      *(unsigned int*)(&sA[cur][aoff[j]]) = pA[j];
#endif
#pragma unroll
    for (int j = 0; j < 8; ++j)
      *(unsigned int*)(sB + boff[j]) = pB[j];
    __syncthreads();

    if (k0 + 32 < F) {
#if HAS_ASYNC_LDS
#pragma unroll
      for (int j = 0; j < 4; ++j)
        async_copy_b32(hbuf + agoff[j] + k0 + 32, &sA[cur ^ 1][aoff[j]]);
#else
#pragma unroll
      for (int j = 0; j < 4; ++j)
        pA[j] = *(const unsigned int*)(hbuf + agoff[j] + k0 + 32);
#endif
#pragma unroll
      for (int j = 0; j < 8; ++j) {
        int o = bgoff[j] + (k0 + 32) * H;
        pB[j] = f2bf_pk(Wd[o], Wd[o + H]);
      }
    }

    v16bf a[2], b[2];
#pragma unroll
    for (int i = 0; i < 2; ++i)
      a[i] = *(const v16bf*)(&sA[cur][(((wm << 1) + i) << 9) + (lane << 4)]);
#pragma unroll
    for (int j = 0; j < 2; ++j)
      b[j] = *(const v16bf*)(sB + (((wn << 1) + j) << 9) + (lane << 4));
#pragma unroll
    for (int i = 0; i < 2; ++i)
#pragma unroll
      for (int j = 0; j < 2; ++j)
        acc[i][j] = __builtin_amdgcn_wmma_f32_16x16x32_bf16(
            false, a[i], false, b[j], (short)0, acc[i][j], false, false);
    __syncthreads();
  }

#pragma unroll
  for (int i = 0; i < 2; ++i) {
#pragma unroll
    for (int j = 0; j < 2; ++j) {
#pragma unroll
      for (int r = 0; r < 8; ++r) {
        int row = (wm << 5) + (i << 4) + r + ((lane >> 4) << 3);
        int slot = m0 + row;
        if (slot < cnt) {
          int tokn = idx_e[slot];
          float w = wgt_e[slot];
          int col = n0 + (wn << 5) + (j << 4) + (lane & 15);
          float* po = out + (size_t)tokn * H + col;
          *po += acc[i][j][r] * w;
        }
      }
    }
  }
}

// ---------------------------------------------------------------------------
// Host launch
// ---------------------------------------------------------------------------
extern "C" void kernel_launch(void* const* d_in, const int* in_sizes, int n_in,
                              void* d_out, int out_size, void* d_ws,
                              size_t ws_size, hipStream_t stream) {
  (void)in_sizes; (void)n_in; (void)out_size; (void)ws_size;

  const float* x   = (const float*)d_in[0];  // [T, H]
  const float* wg  = (const float*)d_in[1];  // [H, E]
  const float* Wgp = (const float*)d_in[2];  // [E, H, F]
  const float* Wup = (const float*)d_in[3];  // [E, H, F]
  const float* Wdp = (const float*)d_in[4];  // [E, F, H]
  float* out = (float*)d_out;                // [T, H]

  // Workspace layout (~42 MB)
  char* ws = (char*)d_ws;
  unsigned short* xb   = (unsigned short*)ws;                        // T*H bf16
  unsigned short* hbuf = (unsigned short*)(ws + (size_t)T * H * 2);  // T*F bf16
  char* p2 = ws + (size_t)T * H * 2 + (size_t)T * F * 2;
  int*   idx = (int*)p2;                          // E*T
  float* wgt = (float*)(p2 + (size_t)E * T * 4);  // E*T
  int*   cnt = (int*)(p2 + (size_t)E * T * 8);    // E

  (void)hipMemsetAsync(cnt, 0, E * sizeof(int), stream);
  (void)hipMemsetAsync(out, 0, (size_t)T * H * sizeof(float), stream);

  {
    int n2 = T * H / 2;
    cvt_bf16_kernel<<<(n2 + 255) / 256, 256, 0, stream>>>(
        x, (unsigned int*)xb, n2);
  }
  moe_router_kernel<<<(T * 32) / 256, 256, 0, stream>>>(x, wg, idx, wgt, cnt);

  for (int e = 0; e < E; ++e) {
    const float* Wg_e = Wgp + (size_t)e * H * F;
    const float* Wu_e = Wup + (size_t)e * H * F;
    const float* Wd_e = Wdp + (size_t)e * F * H;
    const int*   idx_e = idx + (size_t)e * T;
    const float* wgt_e = wgt + (size_t)e * T;
    const int*   cnt_e = cnt + e;

    moe_gate_up_kernel<<<dim3(F / 128, T / 64), 256, 0, stream>>>(
        xb, Wg_e, Wu_e, idx_e, cnt_e, hbuf);
    moe_down_kernel<<<dim3(H / 128, T / 64), 256, 0, stream>>>(
        hbuf, Wd_e, idx_e, wgt_e, cnt_e, out);
  }
}
